// nUIV_NODE_64037962383661
// MI455X (gfx1250) — matrix operand order repaired and checked by reading, not compiled, not run on hardware
//
#include <hip/hip_runtime.h>

#define NN 8192

typedef __attribute__((ext_vector_type(16))) _Float16 v16h;
typedef __attribute__((ext_vector_type(8)))  _Float16 v8h;
typedef __attribute__((ext_vector_type(8)))  float    v8f;

// ---------------------------------------------------------------------------
// Init: abs() of params/state, diag(ts), f16 copy of V, zero-free everywhere.
// ---------------------------------------------------------------------------
__global__ __launch_bounds__(256)
void k_init(const float* x0, const float* betas, const float* deltas,
            const float* cs, const float* ps, const float* ts,
            float* U0, float* I0, float* V0, _Float16* VhS,
            float* beta, float* delta, float* pp, float* cc, float* diag)
{
    int n = blockIdx.x * blockDim.x + threadIdx.x;
    if (n < NN) {
        float u = fabsf(x0[3 * n + 0]);
        float i = fabsf(x0[3 * n + 1]);
        float v = fabsf(x0[3 * n + 2]);
        U0[n] = u; I0[n] = i; V0[n] = v;
        VhS[n] = (_Float16)v;
        beta[n]  = fabsf(betas[n]);
        delta[n] = fabsf(deltas[n]);
        pp[n]    = fabsf(ps[n]);
        cc[n]    = fabsf(cs[n]);
        diag[n]  = ts[(size_t)n * NN + n];
    }
}

// ---------------------------------------------------------------------------
// Transpose ts (fp32 row-major) -> tsT (f16, pre-scaled by 2^13 so the
// uniform(0,1)/N entries land in the f16 normal range). Coalesced via LDS.
// ---------------------------------------------------------------------------
__global__ __launch_bounds__(256)
void k_transpose(const float* __restrict__ ts, _Float16* __restrict__ tsT)
{
    __shared__ float tile[32][33];
    const int tx = threadIdx.x, ty = threadIdx.y;
    const int bi = blockIdx.x,  bj = blockIdx.y;
#pragma unroll
    for (int kk = 0; kk < 4; ++kk) {
        int ry = ty + kk * 8;
        tile[ry][tx] = ts[(size_t)(bj * 32 + ry) * NN + bi * 32 + tx];
    }
    __syncthreads();
#pragma unroll
    for (int kk = 0; kk < 4; ++kk) {
        int ry = ty + kk * 8;
        tsT[(size_t)(bi * 32 + ry) * NN + bj * 32 + tx] =
            (_Float16)(tile[tx][ry] * 8192.0f);
    }
}

// ---------------------------------------------------------------------------
// Fused RK4 stage: WMMA GEMV (coupling = tsT @ Vstage) + per-node RHS +
// stage-vector update. 4 waves/block, each wave owns a 16-row output tile.
//   A (16x32 f16)  : tsT tile, ISA A-layout (K {0..7,16..23} / {8..15,24..31})
//   B (32x16 f16)  : V segment broadcast across all 16 columns
//   C (16x16 f32)  : fp32 accumulator across K = 8192
// ---------------------------------------------------------------------------
template <int STAGE>
__global__ __launch_bounds__(128)
void k_stage(const _Float16* __restrict__ tsT, const _Float16* __restrict__ vh,
             const float* Uin, const float* Iin, const float* Vin,
             const float* U0, const float* I0, const float* V0,
             float* kaU, float* kaI, float* kaV,
             const float* beta, const float* delta, const float* pp,
             const float* cc, const float* diag,
             const float* times, int step,
             float* Uout, float* Iout, float* Vout, _Float16* vhOut)
{
    __shared__ float sCoup[64];

    const int tid  = threadIdx.x;
    const int wave = tid >> 5;
    const int lane = tid & 31;
    const int half = lane >> 4;        // 0: lanes 0-15, 1: lanes 16-31
    const int lr   = lane & 15;
    const int row0 = blockIdx.x * 64 + wave * 16;
    const _Float16* arow = tsT + (size_t)(row0 + lr) * NN;
    const int h8  = half * 8;
    const int h16 = half * 16;

    v8f c = {0.f, 0.f, 0.f, 0.f, 0.f, 0.f, 0.f, 0.f};

#pragma unroll 4
    for (int j0 = 0; j0 < NN; j0 += 32) {
        v16h a, b;
        // A: row (row0+lr), K-chunks per ISA 16-bit A layout
        *(v8h*)&a        = *(const v8h*)(arow + j0 + h8);
        *((v8h*)&a + 1)  = *(const v8h*)(arow + j0 + 16 + h8);
        // B: V K-chunks {h8..h8+7, 16+h8..16+h8+7} broadcast across the 16 N-columns
        *(v8h*)&b        = *(const v8h*)(vh + j0 + h8);
        *((v8h*)&b + 1)  = *(const v8h*)(vh + j0 + 16 + h8);
        c = __builtin_amdgcn_wmma_f32_16x16x32_f16(
                false, a, false, b, (short)0, c, false, false);
    }

    // All 16 columns of D are identical; lanes 0 and 16 of each wave hold
    // rows M = r and M = r + 8 respectively (C/D layout, ISA 7.12.2).
    if (lr == 0) {
        const float inv = 1.0f / 8192.0f;   // undo the 2^13 pre-scale
#pragma unroll
        for (int r = 0; r < 8; ++r)
            sCoup[wave * 16 + half * 8 + r] = c[r] * inv;
    }
    __syncthreads();

    if (tid < 64) {
        const int n = blockIdx.x * 64 + tid;
        const float h    = times[step + 1] - times[step];
        const float coup = sCoup[tid];
        const float Ui = Uin[n], Ii = Iin[n], Vi = Vin[n];
        const float infv = beta[n] * Ui * Vi;
        const float kU = -infv;
        const float kI = infv - delta[n] * Ii;
        const float kV = pp[n] * Ii - cc[n] * Vi + coup - diag[n] * Vi;

        float uo, io, vo;
        if (STAGE == 1) {
            kaU[n] = kU; kaI[n] = kI; kaV[n] = kV;
            const float a = 0.5f * h;
            uo = U0[n] + a * kU; io = I0[n] + a * kI; vo = V0[n] + a * kV;
        } else if (STAGE == 2) {
            kaU[n] += 2.0f * kU; kaI[n] += 2.0f * kI; kaV[n] += 2.0f * kV;
            const float a = 0.5f * h;
            uo = U0[n] + a * kU; io = I0[n] + a * kI; vo = V0[n] + a * kV;
        } else if (STAGE == 3) {
            kaU[n] += 2.0f * kU; kaI[n] += 2.0f * kI; kaV[n] += 2.0f * kV;
            uo = U0[n] + h * kU; io = I0[n] + h * kI; vo = V0[n] + h * kV;
        } else {
            const float s = h * (1.0f / 6.0f);
            uo = U0[n] + s * (kaU[n] + kU);
            io = I0[n] + s * (kaI[n] + kI);
            vo = V0[n] + s * (kaV[n] + kV);
        }
        Uout[n] = uo; Iout[n] = io; Vout[n] = vo;
        vhOut[n] = (_Float16)vo;
    }
}

// ---------------------------------------------------------------------------
// Readout: SIR[c][t] = (1/N) * sum_n sigmoid(slope[c]*(W[c,:]@uiv[n]+b[c]-thr[c]))
// Single block, deterministic tree reduction.
// ---------------------------------------------------------------------------
__global__ __launch_bounds__(256)
void k_readout(const float* U0, const float* I0, const float* V0,
               const float* W, const float* b, const float* slope,
               const float* thr, float* out, int t, int T)
{
    __shared__ float red[3][256];
    float a0 = 0.f, a1 = 0.f, a2 = 0.f;
    for (int n = threadIdx.x; n < NN; n += 256) {
        float u = U0[n], i = I0[n], v = V0[n];
        float l0 = W[0] * u + W[1] * i + W[2] * v + b[0];
        float l1 = W[3] * u + W[4] * i + W[5] * v + b[1];
        float l2 = W[6] * u + W[7] * i + W[8] * v + b[2];
        a0 += 1.0f / (1.0f + __expf(-slope[0] * (l0 - thr[0])));
        a1 += 1.0f / (1.0f + __expf(-slope[1] * (l1 - thr[1])));
        a2 += 1.0f / (1.0f + __expf(-slope[2] * (l2 - thr[2])));
    }
    red[0][threadIdx.x] = a0; red[1][threadIdx.x] = a1; red[2][threadIdx.x] = a2;
    __syncthreads();
    for (int s = 128; s > 0; s >>= 1) {
        if (threadIdx.x < (unsigned)s) {
            red[0][threadIdx.x] += red[0][threadIdx.x + s];
            red[1][threadIdx.x] += red[1][threadIdx.x + s];
            red[2][threadIdx.x] += red[2][threadIdx.x + s];
        }
        __syncthreads();
    }
    if (threadIdx.x == 0) {
        const float inv = 1.0f / (float)NN;
        out[0 * T + t] = red[0][0] * inv;
        out[1 * T + t] = red[1][0] * inv;
        out[2 * T + t] = red[2][0] * inv;
    }
}

// ---------------------------------------------------------------------------
extern "C" void kernel_launch(void* const* d_in, const int* in_sizes, int n_in,
                              void* d_out, int out_size, void* d_ws, size_t ws_size,
                              hipStream_t stream)
{
    (void)n_in; (void)out_size; (void)ws_size;
    const float* times  = (const float*)d_in[0];
    const float* x0     = (const float*)d_in[1];
    const float* betas  = (const float*)d_in[2];
    const float* deltas = (const float*)d_in[3];
    const float* cs     = (const float*)d_in[4];
    const float* ps     = (const float*)d_in[5];
    const float* ts     = (const float*)d_in[6];
    const float* W      = (const float*)d_in[7];
    const float* b      = (const float*)d_in[8];
    const float* slope  = (const float*)d_in[9];
    const float* thr    = (const float*)d_in[10];
    float* out = (float*)d_out;
    const int T = in_sizes[0];

    // ---- workspace carve: 128 MB f16 tsT + small state/param arrays ----
    char* w = (char*)d_ws;
    _Float16* tsT = (_Float16*)w;  w += (size_t)NN * NN * sizeof(_Float16);
    auto fa = [&]() { float* p = (float*)w; w += NN * sizeof(float); return p; };
    float *U0 = fa(), *I0 = fa(), *V0 = fa();
    float *s2U = fa(), *s2I = fa(), *s2V = fa();
    float *s3U = fa(), *s3I = fa(), *s3V = fa();
    float *s4U = fa(), *s4I = fa(), *s4V = fa();
    float *kaU = fa(), *kaI = fa(), *kaV = fa();
    float *beta = fa(), *delta = fa(), *pp = fa(), *cc = fa(), *diag = fa();
    auto ha = [&]() { _Float16* p = (_Float16*)w; w += NN * sizeof(_Float16); return p; };
    _Float16 *VhS = ha(), *Vh2 = ha(), *Vh3 = ha(), *Vh4 = ha();

    k_init<<<NN / 256, 256, 0, stream>>>(x0, betas, deltas, cs, ps, ts,
                                         U0, I0, V0, VhS,
                                         beta, delta, pp, cc, diag);
    k_transpose<<<dim3(NN / 32, NN / 32), dim3(32, 8), 0, stream>>>(ts, tsT);
    k_readout<<<1, 256, 0, stream>>>(U0, I0, V0, W, b, slope, thr, out, 0, T);

    for (int t = 0; t < T - 1; ++t) {
        k_stage<1><<<NN / 64, 128, 0, stream>>>(tsT, VhS, U0, I0, V0,  U0, I0, V0,
                                                kaU, kaI, kaV, beta, delta, pp, cc, diag,
                                                times, t, s2U, s2I, s2V, Vh2);
        k_stage<2><<<NN / 64, 128, 0, stream>>>(tsT, Vh2, s2U, s2I, s2V, U0, I0, V0,
                                                kaU, kaI, kaV, beta, delta, pp, cc, diag,
                                                times, t, s3U, s3I, s3V, Vh3);
        k_stage<3><<<NN / 64, 128, 0, stream>>>(tsT, Vh3, s3U, s3I, s3V, U0, I0, V0,
                                                kaU, kaI, kaV, beta, delta, pp, cc, diag,
                                                times, t, s4U, s4I, s4V, Vh4);
        k_stage<4><<<NN / 64, 128, 0, stream>>>(tsT, Vh4, s4U, s4I, s4V, U0, I0, V0,
                                                kaU, kaI, kaV, beta, delta, pp, cc, diag,
                                                times, t, U0, I0, V0, VhS);
        k_readout<<<1, 256, 0, stream>>>(U0, I0, V0, W, b, slope, thr, out, t + 1, T);
    }
}
